// AudioVisualModel_62569083568612
// MI455X (gfx1250) — compile-verified, matrix operand order checked by reading, not run on hardware
//
#include <hip/hip_runtime.h>
#include <math.h>

// ---- problem constants (from reference) ----
#define BB   16
#define NA   128
#define TT   10
#define NV   196
#define DD   256
#define MROWS (BB*NA)          // 2048 audio rows
#define VROWS (BB*TT*NV)       // 31360 visual rows
#define NTILES 13              // ceil(196/16)
#define TOTAL_SIMS 64225280.0f // 16*16*128*10*196
#define NEG_INF (-3.4e38f)
#define LDSW 264               // padded LDS row: 264 halves = 528B -> conflict-free

typedef __attribute__((ext_vector_type(16))) _Float16 v16h;
typedef __attribute__((ext_vector_type(8)))  _Float16 v8h;
typedef __attribute__((ext_vector_type(8)))  float    v8f;
typedef int v4i_t __attribute__((vector_size(16)));   // builtin's b128 unit

#define AS1 __attribute__((address_space(1)))
#define AS3 __attribute__((address_space(3)))

#if defined(__has_builtin)
#  if __has_builtin(__builtin_amdgcn_global_load_async_to_lds_b128)
#    define HAVE_ASYNC_LDS 1
#  endif
#endif

static __device__ inline void wait_async_done() {
#ifdef HAVE_ASYNC_LDS
#  if __has_builtin(__builtin_amdgcn_s_wait_asynccnt)
    __builtin_amdgcn_s_wait_asynccnt(0);
#  else
    asm volatile("s_wait_asynccnt 0x0" ::: "memory");
#  endif
#endif
}

// Load one 16x16x32-f16 WMMA operand fragment for this lane (works for
// global A rows and LDS-staged B rows).  ISA layout: lanes 0-15 take
// K {0..7, 16..23}, lanes 16-31 take K {8..15, 24..31}. Two b128 loads.
static __device__ inline v16h load_frag(const _Float16* p) {
    v8h lo = *(const v8h*)(p);
    v8h hi = *(const v8h*)(p + 16);
    v16h r;
#pragma unroll
    for (int i = 0; i < 8; ++i) { r[i] = lo[i]; r[i + 8] = hi[i]; }
    return r;
}

// ---- kernel 0: zero the accumulator block (clip[256] + nonneg[1]) ----
__global__ void avk_zero(float* acc) {
    acc[threadIdx.x] = 0.0f;
    if (threadIdx.x == 0) acc[256] = 0.0f;
}

// ---- kernel 1: row-wise L2 normalize fp32 -> f16 (one block per row) ----
__global__ void avk_l2norm(const float* __restrict__ in,
                           _Float16* __restrict__ out, int rows) {
    int row = blockIdx.x;
    if (row >= rows) return;
    __shared__ float red[DD];
    int t = threadIdx.x;
    float x = in[(size_t)row * DD + t];
    red[t] = x * x;
    __syncthreads();
#pragma unroll
    for (int s = DD / 2; s > 0; s >>= 1) {
        if (t < s) red[t] += red[t + s];
        __syncthreads();
    }
    float inv = 1.0f / fmaxf(sqrtf(red[0]), 1e-12f);
    out[(size_t)row * DD + t] = (_Float16)(x * inv);
}

// ---- kernel 2: fused WMMA GEMM + max/nonneg reductions ----
// Block = 128 threads (4 waves) covers 128 audio rows (one batch x) against
// one (y,t) group of 196 patch columns. B tiles staged once into LDS and
// reused by 8 M-tiles (16x reuse vs L2) -> near matrix-bound.
__global__ __launch_bounds__(128) void avk_sim_gemm(
        const _Float16* __restrict__ A,   // (2048, 256) f16, L2-normalized
        const _Float16* __restrict__ V,   // (31360,256) f16, L2-normalized
        float* __restrict__ acc) {        // [0..255]=clip sums, [256]=nonneg
    const int bid  = blockIdx.x;
    const int xb   = bid & (BB - 1);       // audio batch
    const int g    = bid >> 4;             // 0..159
    const int y    = g / TT;
    const int tim  = g - y * TT;

    const int tid  = threadIdx.x;
    const int lane = tid & 31;
    const int w    = tid >> 5;
    const int ml   = lane & 15;            // row-in-tile (A) / col-in-tile (B)
    const int koff = (lane >> 4) * 8;      // K sub-slab per lane half

    __shared__ _Float16 lds_b[2][16][LDSW];     // double-buffered B tile
    __shared__ float    lmax[4][32][16];
    __shared__ float    lred[128];

    // wave w owns block M-tiles w (slot0) and w+4 (slot1): 2x A residency
    const _Float16* arow0 = A + (size_t)(xb * 128 + w * 16 + ml) * DD;
    const _Float16* arow1 = arow0 + (size_t)64 * DD;   // +4 tiles * 16 rows
    v16h afrag0[8], afrag1[8];
#pragma unroll
    for (int kt = 0; kt < 8; ++kt) {
        afrag0[kt] = load_frag(arow0 + kt * 32 + koff);
        afrag1[kt] = load_frag(arow1 + kt * 32 + koff);
    }

    const _Float16* vbase = V + ((size_t)(y * TT + tim) * NV) * DD;

    // cooperative staging of one 16x256 B tile (8KB) into LDS buffer `buf`
    auto stage = [&](int nt, int buf) {
#pragma unroll
        for (int p = 0; p < 4; ++p) {
            int idx   = tid + p * 128;       // 512 chunks of 16B
            int row   = idx >> 5;            // 16 rows
            int chunk = idx & 31;            // 32 x 16B per row
            int nl    = nt * 16 + row;
            int nr    = (nl < NV) ? nl : (NV - 1);
            const _Float16* src = vbase + (size_t)nr * DD + chunk * 8;
            _Float16*       dst = &lds_b[buf][row][chunk * 8];
#ifdef HAVE_ASYNC_LDS
            __builtin_amdgcn_global_load_async_to_lds_b128(
                (AS1 v4i_t*)src, (AS3 v4i_t*)dst, 0, 0);
#else
            *(v8h*)dst = *(const v8h*)src;
#endif
        }
    };

    float nn = 0.0f;
    float rmax0[8], rmax1[8];
#pragma unroll
    for (int r = 0; r < 8; ++r) { rmax0[r] = NEG_INF; rmax1[r] = NEG_INF; }

    stage(0, 0);
    for (int nt = 0; nt < NTILES; ++nt) {
        const int buf = nt & 1;
        wait_async_done();
        __syncthreads();                       // buf staged; prev compute done
        if (nt + 1 < NTILES) stage(nt + 1, buf ^ 1);   // overlap next stage

        const int  nl    = nt * 16 + ml;       // this lane's column
        const bool valid = (nl < NV);
        const _Float16* brow = &lds_b[buf][ml][0];

        v8f c0 = {}, c1 = {};
#pragma unroll
        for (int kt = 0; kt < 8; ++kt) {
            v16h b = load_frag(brow + kt * 32 + koff);   // 2x ds_load_b128
            c0 = __builtin_amdgcn_wmma_f32_16x16x32_f16(
                    false, afrag0[kt], false, b, (short)0, c0, false, false);
            c1 = __builtin_amdgcn_wmma_f32_16x16x32_f16(
                    false, afrag1[kt], false, b, (short)0, c1, false, false);
        }
        // branch-free epilogue: lane holds col nl, rows r+8*(lane>=16)
#pragma unroll
        for (int r = 0; r < 8; ++r) {
            float s0 = c0[r], s1 = c1[r];
            float m0 = valid ? fminf(s0, 0.0f) : 0.0f;
            float m1 = valid ? fminf(s1, 0.0f) : 0.0f;
            nn += m0 * m0 + m1 * m1;
            rmax0[r] = fmaxf(rmax0[r], valid ? s0 : NEG_INF);
            rmax1[r] = fmaxf(rmax1[r], valid ? s1 : NEG_INF);
        }
    }

#pragma unroll
    for (int r = 0; r < 8; ++r) {
        lmax[w][lane][r]     = rmax0[r];
        lmax[w][lane][r + 8] = rmax1[r];
    }
    lred[tid] = nn;
    __syncthreads();

    // block-sum of nonneg partials -> 1 atomic
#pragma unroll
    for (int s = 64; s > 0; s >>= 1) {
        if (tid < s) lred[tid] += lred[tid + s];
        __syncthreads();
    }
    if (tid == 0) atomicAdd(&acc[256], lred[0]);
    __syncthreads();

    // per-row max over the 16 owning lanes, then block-sum -> 1 atomic
    {
        int tileIdx = tid >> 4;                 // block M-tile 0..7
        int M       = tid & 15;                 // row within tile
        int wsrc    = tileIdx & 3;              // owning wave
        int tsel    = tileIdx >> 2;             // which of its 2 slots
        int r       = (M & 7) + tsel * 8;
        int half    = M >> 3;
        float m = NEG_INF;
#pragma unroll
        for (int l = 0; l < 16; ++l)
            m = fmaxf(m, lmax[wsrc][half * 16 + l][r]);
        lred[tid] = m;                          // sum of per-(row,t) maxes
    }
    __syncthreads();
#pragma unroll
    for (int s = 64; s > 0; s >>= 1) {
        if (tid < s) lred[tid] += lred[tid + s];
        __syncthreads();
    }
    if (tid == 0) atomicAdd(&acc[xb * 16 + y], lred[0]);
}

// ---- kernel 3: 16x16 InfoNCE + regularization -> 3 scalars ----
__global__ void avk_finalize(const float* __restrict__ acc,
                             const float* __restrict__ temp,
                             float* __restrict__ out) {
    __shared__ float clip[256];
    int tid = threadIdx.x;
    float tau = *temp;
    clip[tid] = acc[tid] / ((float)(NA * TT) * tau);  // mean over a,t and /tau
    __syncthreads();
    if (tid == 0) {
        float losses = 0.0f;
        for (int i = 0; i < 16; ++i) {
            float mr = NEG_INF, mc = NEG_INF;
            for (int j = 0; j < 16; ++j) {
                mr = fmaxf(mr, clip[i * 16 + j]);
                mc = fmaxf(mc, clip[j * 16 + i]);
            }
            float sr = 0.0f, sc = 0.0f;
            for (int j = 0; j < 16; ++j) {
                sr += expf(clip[i * 16 + j] - mr);
                sc += expf(clip[j * 16 + i] - mc);
            }
            float lpa = clip[i * 16 + i] - (mr + logf(sr));
            float lpv = clip[i * 16 + i] - (mc + logf(sc));
            losses += -(lpa + lpv);
        }
        float contrastive = 0.5f * (losses / 16.0f);
        float l_nonneg = acc[256] / (tau * tau) / TOTAL_SIMS;
        float logt = logf(tau);
        float tl = fmaxf(-logt, 0.0f);             tl = tl * tl * tl * tl;
        float th = fmaxf(logt - logf(3.0f), 0.0f); th = th * th * th * th;
        float reg = l_nonneg + tl + th;
        out[0] = contrastive + 0.3f * reg;
        out[1] = contrastive;
        out[2] = reg;
    }
}

extern "C" void kernel_launch(void* const* d_in, const int* in_sizes, int n_in,
                              void* d_out, int out_size, void* d_ws, size_t ws_size,
                              hipStream_t stream) {
    (void)in_sizes; (void)n_in; (void)out_size; (void)ws_size;
    const float* audio  = (const float*)d_in[0];   // (16,128,256) f32
    const float* visual = (const float*)d_in[1];   // (16,10,196,256) f32
    const float* temp   = (const float*)d_in[2];   // scalar f32
    float* out = (float*)d_out;                    // 3 floats

    // workspace layout: [0,2048) accumulators, then f16 normalized tensors
    float*    acc = (float*)d_ws;
    _Float16* An  = (_Float16*)((char*)d_ws + 2048);
    _Float16* Vn  = An + (size_t)MROWS * DD;       // +1 MB
    // Vn needs VROWS*DD*2 = ~16 MB; total ws use ~17.1 MB

    avk_zero<<<1, 256, 0, stream>>>(acc);
    avk_l2norm<<<MROWS, DD, 0, stream>>>(audio, An, MROWS);
    avk_l2norm<<<VROWS, DD, 0, stream>>>(visual, Vn, VROWS);
    avk_sim_gemm<<<BB * BB * TT, 128, 0, stream>>>(An, Vn, acc);
    avk_finalize<<<1, 256, 0, stream>>>(acc, temp, out);
}
